// TTLSTMFullCell_35845797053077
// MI455X (gfx1250) — matrix operand order repaired and checked by reading, not compiled, not run on hardware
//
#include <hip/hip_runtime.h>

// CDNA5 (gfx1250) wave32 WMMA types
typedef __bf16   v16bf __attribute__((ext_vector_type(16)));
typedef float    v8f   __attribute__((ext_vector_type(8)));
typedef float    v4f   __attribute__((ext_vector_type(4)));
typedef unsigned v4u   __attribute__((ext_vector_type(4)));
typedef unsigned v2u   __attribute__((ext_vector_type(2)));

union Frag { v16bf v; unsigned u[8]; v4u q[2]; };

// ---------------------------------------------------------------------------
// Setup: merge TT cores pairwise.
//   Mw[row'=(o2,o3)*16+r2][col=(i2,i3)] = sum_r3 core2[(r2,o2),(i2,r3)] * core3[(r3,o3),i3]
//     (rows permuted so GEMM1 D-tiles scatter to CONTIGUOUS a2t addresses)
//   Nw[(o0',o1)=64][(i0,i1,r2)=1024]   = sum_r1 core0[o0',(i0,r1)] * core1[(r1,o1),(i1,r2)]
// ---------------------------------------------------------------------------
__global__ void tt_merge_cores(const float* __restrict__ core0,
                               const float* __restrict__ core1,
                               const float* __restrict__ core2,
                               const float* __restrict__ core3,
                               __bf16* __restrict__ Mw,
                               __bf16* __restrict__ Nw) {
  int idx = blockIdx.x * blockDim.x + threadIdx.x;
  if (idx < 65536) {
    int row = idx >> 6, col = idx & 63;
    int o2 = (row >> 7) & 7, o3 = (row >> 4) & 7, r2 = row & 15;
    int i2 = col >> 3, i3 = col & 7;
    float s = 0.f;
    for (int r3 = 0; r3 < 16; ++r3)
      s += core2[(r2 * 8 + o2) * 128 + i2 * 16 + r3] *
           core3[(r3 * 8 + o3) * 8 + i3];
    Mw[idx] = (__bf16)s;
  } else if (idx < 131072) {
    int j = idx - 65536;
    int row = j >> 10, col = j & 1023;
    int o0 = row >> 2, o1 = row & 3;
    int i0 = col >> 7, i1 = (col >> 4) & 7, r2 = col & 15;
    float s = 0.f;
    for (int r1 = 0; r1 < 16; ++r1)
      s += core0[o0 * 128 + i0 * 16 + r1] *
           core1[(r1 * 4 + o1) * 128 + i1 * 16 + r2];
    Nw[j] = (__bf16)s;
  }
}

// ---------------------------------------------------------------------------
// Persistent sequential LSTM: one workgroup (4 wave32) per batch sample.
// Per timestep: Z = N @ regroup( M @ Xr ), gates, state update. All in LDS.
// x_{t+1} streams memory->LDS via GLOBAL_LOAD_ASYNC_TO_LDS_B128 (ASYNCcnt),
// overlapping with the WMMAs of step t.
// ---------------------------------------------------------------------------
#define THREADS 128

// LDS carve (bytes)
#define OFF_XV    0        // 4096 bf16  concat [x_t | h]            (8 KB)
#define OFF_A2T   8192     // 64x1024 bf16 GEMM2 B operand           (128 KB)
#define OFF_ZBUF  139264   // 4096 f32  pre-activations (transposed) (16 KB)
#define OFF_CBUF  155648   // 1024 f32  cell state                   (4 KB)
#define OFF_XS    159744   // 2 x 3072 f32 async staging             (24 KB)
#define SMEM_SZ   184320   // 180 KB < 320 KB WGP LDS

__device__ __forceinline__ void async_x_to_lds(const float4* xp, float* dstf, int tid) {
  #pragma unroll
  for (int j = tid; j < 768; j += THREADS) {
    unsigned dst = (unsigned)(unsigned long long)(dstf + 4 * j);
    unsigned long long src = (unsigned long long)(xp + j);
    asm volatile("global_load_async_to_lds_b128 %0, %1, off"
                 :: "v"(dst), "v"(src) : "memory");
  }
}

__global__ void __launch_bounds__(THREADS)
tt_lstm_seq(const float* __restrict__ x,
            const __bf16* __restrict__ Mw,
            const __bf16* __restrict__ Nw,
            const float* __restrict__ bias,
            float* __restrict__ out) {
  extern __shared__ unsigned char smem[];
  __bf16* xv   = (__bf16*)(smem + OFF_XV);
  __bf16* a2t  = (__bf16*)(smem + OFF_A2T);
  float*  zbuf = (float*)(smem + OFF_ZBUF);
  float*  cbuf = (float*)(smem + OFF_CBUF);
  float*  xs   = (float*)(smem + OFF_XS);

  const int b    = blockIdx.x;        // one sample per workgroup
  const int tid  = threadIdx.x;
  const int wave = tid >> 5;
  const int lane = tid & 31;
  const int lo   = lane & 15, hi = lane >> 4;

  const unsigned* MwU = (const unsigned*)Mw;    // row stride 32 uints (64 bf16)
  const unsigned* NwU = (const unsigned*)Nw;    // row stride 512 uints (1024 bf16)
  const unsigned* xvU = (const unsigned*)xv;    // row stride 32 uints
  const unsigned* a2U = (const unsigned*)a2t;   // row stride 512 uints

  for (int h = tid; h < 1024; h += THREADS) { cbuf[h] = 0.f; xv[3072 + h] = (__bf16)0.f; }

  // pre-issue async load of x_0[b] into staging buffer 0
  async_x_to_lds((const float4*)(x + (size_t)b * 3072), xs, tid);

  for (int t = 0; t < 256; ++t) {
    // ---- wait for x_t staging (own wave), then make it WG-visible
    asm volatile("s_wait_asynccnt 0" ::: "memory");
    __syncthreads();

    // ---- convert staged f32 -> bf16 concat buffer (packed v_cvt + b64 stores)
    const float4* cur = (const float4*)(xs + (t & 1) * 3072);
    #pragma unroll
    for (int j = tid; j < 768; j += THREADS) {
      float4 v4 = cur[j];
      union { __bf16 h[4]; v2u q; } c4;
      c4.h[0] = (__bf16)v4.x; c4.h[1] = (__bf16)v4.y;
      c4.h[2] = (__bf16)v4.z; c4.h[3] = (__bf16)v4.w;
      *(v2u*)(xv + 4 * j) = c4.q;
    }
    // ---- overlap: stream x_{t+1} into the other staging buffer
    if (t < 255)
      async_x_to_lds((const float4*)(x + ((size_t)(t + 1) * 128 + b) * 3072),
                     xs + ((t + 1) & 1) * 3072, tid);
    __syncthreads();

    // ---- GEMM1: A2 = M[1024x64] @ Xr[64x64]  (K = (i2,i3) = 64)
    // Xr stored transposed in LDS as xv[(i0,i1)][(i2,i3)] == natural concat order.
    Frag bf[2][4];   // all 8 B fragments preloaded, reused by all row tiles
    #pragma unroll
    for (int kk = 0; kk < 2; ++kk)
      #pragma unroll
      for (int nt = 0; nt < 4; ++nt) {
        const v4u* rp = (const v4u*)(xvU + (nt * 16 + lo) * 32 + kk * 16 + 4 * hi);
        bf[kk][nt].q[0] = rp[0];
        bf[kk][nt].q[1] = rp[2];
      }
    for (int rt = wave; rt < 64; rt += 4) {
      Frag a0, a1;
      const unsigned* ar = MwU + (rt * 16 + lo) * 32;
      a0.q[0] = *(const v4u*)(ar + 4 * hi);        // K 0..7 / 8..15
      a0.q[1] = *(const v4u*)(ar + 8 + 4 * hi);    // K 16..23 / 24..31
      a1.q[0] = *(const v4u*)(ar + 16 + 4 * hi);
      a1.q[1] = *(const v4u*)(ar + 24 + 4 * hi);
      #pragma unroll
      for (int nt = 0; nt < 4; ++nt) {
        v8f acc = {};
        acc = __builtin_amdgcn_wmma_f32_16x16x32_bf16(false, a0.v, false, bf[0][nt].v,
                                                      (short)0, acc, false, false);
        acc = __builtin_amdgcn_wmma_f32_16x16x32_bf16(false, a1.v, false, bf[1][nt].v,
                                                      (short)0, acc, false, false);
        // D rows are (o2,o3)=rt, r2 = v + 8*hi  ->  contiguous bf16 in a2t
        union { __bf16 h[8]; v4u q; } pk;
        #pragma unroll
        for (int v = 0; v < 8; ++v) pk.h[v] = (__bf16)acc[v];
        *(v4u*)(a2t + rt * 1024 + (nt * 16 + lo) * 16 + 8 * hi) = pk.q;
      }
    }
    __syncthreads();

    // ---- GEMM2: Z[64x64] = N[64x1024] @ A2r[1024x64]  (each wave: one 16-row stripe)
    {
      v8f zacc[4] = {};
      const int zr = wave;
      for (int kk = 0; kk < 32; ++kk) {
        Frag a;
        const unsigned* ar = NwU + (zr * 16 + lo) * 512 + kk * 16;
        a.q[0] = *(const v4u*)(ar + 4 * hi);
        a.q[1] = *(const v4u*)(ar + 8 + 4 * hi);
        #pragma unroll
        for (int nt = 0; nt < 4; ++nt) {
          Frag bb;
          const v4u* bp = (const v4u*)(a2U + (nt * 16 + lo) * 512 + kk * 16 + 4 * hi);
          bb.q[0] = bp[0];
          bb.q[1] = bp[2];
          zacc[nt] = __builtin_amdgcn_wmma_f32_16x16x32_bf16(false, a.v, false, bb.v,
                                                             (short)0, zacc[nt], false, false);
        }
      }
      // store Z transposed: zbuf[col*64 + row] -> 8 contiguous f32 per lane
      #pragma unroll
      for (int nt = 0; nt < 4; ++nt) {
        union { v8f v; v4f f4[2]; } zp;
        zp.v = zacc[nt];
        float* zd = zbuf + (nt * 16 + lo) * 64 + zr * 16 + 8 * hi;
        *(v4f*)zd = zp.f4[0];
        *(v4f*)(zd + 4) = zp.f4[1];
      }
    }
    __syncthreads();

    // ---- gates + state update; h feeds back into the concat buffer
    float* outp = out + ((size_t)t * 128 + b) * 1024;
    for (int h = tid; h < 1024; h += THREADS) {
      // feature f = g*1024 + h; zbuf is transposed: z = zbuf[(f&63)*64 + (f>>6)]
      int f0 = h, f1 = 1024 + h, f2 = 2048 + h, f3 = 3072 + h;
      float z0 = zbuf[(f0 & 63) * 64 + (f0 >> 6)] + bias[f0];
      float z1 = zbuf[(f1 & 63) * 64 + (f1 >> 6)] + bias[f1];
      float z2 = zbuf[(f2 & 63) * 64 + (f2 >> 6)] + bias[f2];
      float z3 = zbuf[(f3 & 63) * 64 + (f3 >> 6)] + bias[f3];
      float ig = fminf(fmaxf(z0 * (1.f / 6.f) + 0.5f, 0.f), 1.f);
      float fg = fminf(fmaxf(z1 * (1.f / 6.f) + 0.5f, 0.f), 1.f);
      float og = fminf(fmaxf(z3 * (1.f / 6.f) + 0.5f, 0.f), 1.f);
      float cn = fg * cbuf[h] + ig * tanhf(z2);
      cbuf[h] = cn;
      float hn = og * tanhf(cn);
      outp[h] = hn;
      xv[3072 + h] = (__bf16)hn;
    }
    __syncthreads();
  }
}

// ---------------------------------------------------------------------------
extern "C" void kernel_launch(void* const* d_in, const int* in_sizes, int n_in,
                              void* d_out, int out_size, void* d_ws, size_t ws_size,
                              hipStream_t stream) {
  (void)in_sizes; (void)n_in; (void)out_size; (void)ws_size;
  const float* x     = (const float*)d_in[0];
  const float* core0 = (const float*)d_in[1];
  const float* core1 = (const float*)d_in[2];
  const float* core2 = (const float*)d_in[3];
  const float* core3 = (const float*)d_in[4];
  const float* bias  = (const float*)d_in[5];

  __bf16* Mw = (__bf16*)d_ws;   // 1024x64 bf16, rows permuted (o2,o3,r2)
  __bf16* Nw = Mw + 65536;      // 64x1024 bf16

  tt_merge_cores<<<512, 256, 0, stream>>>(core0, core1, core2, core3, Mw, Nw);

  tt_lstm_seq<<<128, THREADS, SMEM_SZ, stream>>>(x, Mw, Nw, bias, (float*)d_out);
}
